// HNet_29197187678861
// MI455X (gfx1250) — matrix operand order repaired and checked by reading, not compiled, not run on hardware
//
#include <hip/hip_runtime.h>
#include <hip/hip_bf16.h>
#include <math.h>

// ---------------------------------------------------------------------------
// H-Net forward on MI455X (gfx1250, wave32).
// All matmuls go through v_wmma_f32_16x16x32_bf16 (bf16 A/B, fp32 accum).
// Block tile 64x64: 4 waves, each wave owns a 16-row M-slice and computes all
// four 16-col N-subtiles (4 accumulators), reusing one register-resident A
// fragment across 4 WMMAs per K-step. B tiles are cooperatively staged to LDS
// (wide unpredicated float4 loads on full tiles, bf16-converted) and shared by
// all waves via ds_load_b128. A/B use contiguous k = 16*(lane/16) + s.
// ---------------------------------------------------------------------------

typedef __attribute__((ext_vector_type(16))) __bf16 v16bf;
typedef __attribute__((ext_vector_type(8)))  float  v8f;

#define GEMM_BIAS   1
#define GEMM_RES    2
#define GEMM_GELU   4
#define GEMM_TRANSB 8

__device__ __forceinline__ float gelu_f(float x) {
  const float kk = 0.7978845608028654f;  // sqrt(2/pi)
  float x3 = x * x * x;
  return 0.5f * x * (1.0f + tanhf(kk * (x + 0.044715f * x3)));
}

// Generic batched-strided WMMA GEMM: C[M,N] = alpha*A[M,K]@B[K,N] (+bias)(+gelu)(+res)
// grid = (ceil(N/64), ceil(M/64), batch*nh); block = 128 (4 waves).
__global__ void __launch_bounds__(128) wmma_gemm_kernel(
    const float* __restrict__ A, const float* __restrict__ Bm,
    const float* __restrict__ bias, const float* __restrict__ Res,
    float* __restrict__ C,
    int M, int N, int K, int lda, int ldb, int ldc,
    long long sAb, long long sAh, long long sBb, long long sBh,
    long long sCb, long long sCh, int nh, float alpha, int flags)
{
  // B tile staged [n][k] as bf16; row stride 40 keeps 16B alignment and spreads banks
  __shared__ __bf16 Bs[64][40];

  const int wave = threadIdx.x >> 5;
  const int lane = threadIdx.x & 31;
  const int nb0  = (int)blockIdx.x * 64;
  const int m0   = (int)blockIdx.y * 64 + wave * 16;   // per-wave M slice
  const int z    = (int)blockIdx.z;
  const int bb = z / nh;
  const int hh = z - bb * nh;
  const float* Ap = A  + (long long)bb * sAb + (long long)hh * sAh;
  const float* Bp = Bm + (long long)bb * sBb + (long long)hh * sBh;
  float*       Cp = C  + (long long)bb * sCb + (long long)hh * sCh;
  const float* Rp = Res ? (Res + (long long)bb * sCb + (long long)hh * sCh) : (const float*)0;

  const int  sub = lane & 15;
  const bool hi  = lane >= 16;
  int arow = m0 + sub; if (arow >= M) arow = M - 1;   // clamped rows never stored

  const bool transb = (flags & GEMM_TRANSB) != 0;
  const bool fullN  = (nb0 + 64 <= N);                 // block-uniform

  v8f acc[4];
#pragma unroll
  for (int nt = 0; nt < 4; ++nt) acc[nt] = (v8f){0.f, 0.f, 0.f, 0.f, 0.f, 0.f, 0.f, 0.f};

  const int kFull = K & ~31;

  for (int kb = 0; kb < kFull; kb += 32) {
    // ---- cooperative staging of B tile (32 K x 64 N) into LDS as bf16 ----
    __syncthreads();            // protect previous iteration's fragment reads
    if (transb) {
      if (fullN) {
        // B stored [N][K]: float4 along K, packed 8B LDS store (unpredicated)
#pragma unroll
        for (int j = 0; j < 4; ++j) {
          int cid = (int)threadIdx.x + 128 * j;
          int n   = cid >> 3;
          int kc  = (cid & 7) << 2;
          float4 bv = *(const float4*)(Bp + (long long)(nb0 + n) * ldb + kb + kc);
          union { __bf16 h[4]; unsigned long long u; } pk;
          pk.h[0] = (__bf16)bv.x; pk.h[1] = (__bf16)bv.y;
          pk.h[2] = (__bf16)bv.z; pk.h[3] = (__bf16)bv.w;
          *(unsigned long long*)&Bs[n][kc] = pk.u;
        }
      } else {
        for (int l = threadIdx.x; l < 2048; l += 128) {
          int n  = l >> 5;
          int kk = l & 31;
          int gn = nb0 + n;
          float v = (gn < N) ? Bp[(long long)gn * ldb + kb + kk] : 0.0f;
          Bs[n][kk] = (__bf16)v;
        }
      }
    } else {
      if (fullN) {
        // B stored [K][N]: float4 along N (coalesced 256B rows), unpredicated
#pragma unroll
        for (int j = 0; j < 4; ++j) {
          int cid = (int)threadIdx.x + 128 * j;
          int kk  = cid >> 4;
          int nc  = (cid & 15) << 2;
          float4 bv = *(const float4*)(Bp + (long long)(kb + kk) * ldb + nb0 + nc);
          Bs[nc + 0][kk] = (__bf16)bv.x;
          Bs[nc + 1][kk] = (__bf16)bv.y;
          Bs[nc + 2][kk] = (__bf16)bv.z;
          Bs[nc + 3][kk] = (__bf16)bv.w;
        }
      } else {
        for (int l = threadIdx.x; l < 2048; l += 128) {
          int kk = l >> 6;
          int n  = l & 63;
          int gn = nb0 + n;
          float v = (gn < N) ? Bp[(long long)(kb + kk) * ldb + gn] : 0.0f;
          Bs[n][kk] = (__bf16)v;
        }
      }
    }
    __syncthreads();

    // ---- A fragment: lane row, K-run {kb+hi*16 .. +15} ----
    const float* ap = Ap + (long long)arow * lda + kb + (hi ? 16 : 0);
    if (kb + 32 < kFull) __builtin_prefetch(ap + 32, 0, 3);
    float4 a0 = *(const float4*)(ap + 0);
    float4 a1 = *(const float4*)(ap + 4);
    float4 a2 = *(const float4*)(ap + 8);
    float4 a3 = *(const float4*)(ap + 12);
    v16bf af;
    af[0]  = (__bf16)a0.x; af[1]  = (__bf16)a0.y; af[2]  = (__bf16)a0.z; af[3]  = (__bf16)a0.w;
    af[4]  = (__bf16)a1.x; af[5]  = (__bf16)a1.y; af[6]  = (__bf16)a1.z; af[7]  = (__bf16)a1.w;
    af[8]  = (__bf16)a2.x; af[9]  = (__bf16)a2.y; af[10] = (__bf16)a2.z; af[11] = (__bf16)a2.w;
    af[12] = (__bf16)a3.x; af[13] = (__bf16)a3.y; af[14] = (__bf16)a3.z; af[15] = (__bf16)a3.w;

    // ---- 4 WMMAs reusing the A fragment across the 64-wide N tile ----
#pragma unroll
    for (int nt = 0; nt < 4; ++nt) {
      const float4* bls = (const float4*)((const char*)&Bs[nt * 16 + sub][0] + (hi ? 32 : 0));
      union { v16bf v; float4 f[2]; } bu;
      bu.f[0] = bls[0];
      bu.f[1] = bls[1];
      acc[nt] = __builtin_amdgcn_wmma_f32_16x16x32_bf16(false, af, false, bu.v,
                                                        (short)0, acc[nt], false, false);
    }
  }

  // ---- K tail (K % 32 != 0; only the K=16 GEMMs hit this) ----
  if (kFull < K) {
    const int rem = K - kFull;
    __syncthreads();
    if (transb) {
      for (int l = threadIdx.x; l < 2048; l += 128) {
        int n  = l >> 5;
        int kk = l & 31;
        int gn = nb0 + n;
        float v = (gn < N && kk < rem) ? Bp[(long long)gn * ldb + kFull + kk] : 0.0f;
        Bs[n][kk] = (__bf16)v;
      }
    } else {
      for (int l = threadIdx.x; l < 2048; l += 128) {
        int kk = l >> 6;
        int n  = l & 63;
        int gn = nb0 + n;
        float v = (gn < N && kk < rem) ? Bp[(long long)(kFull + kk) * ldb + gn] : 0.0f;
        Bs[n][kk] = (__bf16)v;
      }
    }
    __syncthreads();
    v16bf af;
#pragma unroll
    for (int i = 0; i < 16; ++i) {
      int ka = kFull + i + (hi ? 16 : 0);
      float av = (ka < K) ? Ap[(long long)arow * lda + ka] : 0.0f;
      af[i] = (__bf16)av;
    }
#pragma unroll
    for (int nt = 0; nt < 4; ++nt) {
      const float4* bls = (const float4*)((const char*)&Bs[nt * 16 + sub][0] + (hi ? 32 : 0));
      union { v16bf v; float4 f[2]; } bu;
      bu.f[0] = bls[0];
      bu.f[1] = bls[1];
      acc[nt] = __builtin_amdgcn_wmma_f32_16x16x32_bf16(false, af, false, bu.v,
                                                        (short)0, acc[nt], false, false);
    }
  }

  // ---- epilogue: D layout VGPR j -> row m0+(hi?8:0)+j, col nb0+nt*16+(lane&15) ----
  const int mbase = m0 + (hi ? 8 : 0);
#pragma unroll
  for (int nt = 0; nt < 4; ++nt) {
    int nn = nb0 + nt * 16 + sub;
    if (nn < N) {
      float bval = (flags & GEMM_BIAS) ? bias[nn] : 0.0f;
#pragma unroll
      for (int j = 0; j < 8; ++j) {
        int mm = mbase + j;
        if (mm < M) {
          float v = acc[nt][j] * alpha + bval;
          if (flags & GEMM_GELU) v = gelu_f(v);
          if (flags & GEMM_RES)  v += Rp[(long long)mm * ldc + nn];
          Cp[(long long)mm * ldc + nn] = v;
        }
      }
    }
  }
}

// ---------------- elementwise / reduction kernels ----------------

__global__ void __launch_bounds__(256) ln_kernel(const float* __restrict__ x,
                                                 const float* __restrict__ w,
                                                 float* __restrict__ y, int C)
{
  long long row = blockIdx.x;
  const float* p = x + row * C;
  float* o = y + row * C;
  __shared__ float s1[256], s2[256];
  float a = 0.f, b = 0.f;
  for (int i = threadIdx.x; i < C; i += 256) { float v = p[i]; a += v; b += v * v; }
  s1[threadIdx.x] = a; s2[threadIdx.x] = b; __syncthreads();
  for (int st = 128; st > 0; st >>= 1) {
    if (threadIdx.x < st) { s1[threadIdx.x] += s1[threadIdx.x + st]; s2[threadIdx.x] += s2[threadIdx.x + st]; }
    __syncthreads();
  }
  float mean = s1[0] / C;
  float var  = s2[0] / C - mean * mean;
  float sc = rsqrtf(var + 1e-5f);
  for (int i = threadIdx.x; i < C; i += 256) o[i] = (p[i] - mean) * sc * w[i];
}

__global__ void __launch_bounds__(256) softmax_kernel(float* __restrict__ x, int N, int t, int causal)
{
  long long row = blockIdx.x;
  float* p = x + row * N;
  int q = causal ? (int)(row % t) : (N + 1);
  __shared__ float s[256];
  float mx = -3.4e38f;
  for (int i = threadIdx.x; i < N; i += 256) {
    float v = (causal && i > q) ? -1e9f : p[i];
    mx = fmaxf(mx, v);
  }
  s[threadIdx.x] = mx; __syncthreads();
  for (int st = 128; st > 0; st >>= 1) {
    if (threadIdx.x < st) s[threadIdx.x] = fmaxf(s[threadIdx.x], s[threadIdx.x + st]);
    __syncthreads();
  }
  mx = s[0]; __syncthreads();
  float sum = 0.f;
  for (int i = threadIdx.x; i < N; i += 256) {
    float v = (causal && i > q) ? -1e9f : p[i];
    float e = __expf(v - mx);
    p[i] = e; sum += e;
  }
  s[threadIdx.x] = sum; __syncthreads();
  for (int st = 128; st > 0; st >>= 1) {
    if (threadIdx.x < st) s[threadIdx.x] += s[threadIdx.x + st];
    __syncthreads();
  }
  float inv = 1.0f / s[0];
  for (int i = threadIdx.x; i < N; i += 256) p[i] *= inv;
}

__global__ void __launch_bounds__(256) embed_kernel(const int* __restrict__ idx,
    const float* __restrict__ tok, const float* __restrict__ pos,
    float* __restrict__ x, int T, int C, long long n)
{
  long long i = (long long)blockIdx.x * 256 + threadIdx.x;
  if (i >= n) return;
  int c = (int)(i % C);
  long long bt = i / C;
  int t = (int)(bt % T);
  x[i] = tok[(long long)idx[bt] * C + c] + pos[(long long)t * C + c];
}

__global__ void __launch_bounds__(256) copy_kernel(float* __restrict__ d,
                                                   const float* __restrict__ s, long long n)
{
  long long i = (long long)blockIdx.x * 256 + threadIdx.x;
  if (i < n) d[i] = s[i];
}

__global__ void __launch_bounds__(256) gate_kernel(const float* __restrict__ x,
    const float* __restrict__ scan, float* __restrict__ gate, int T, int C)
{
  int row = blockIdx.x;           // b*T + t
  int t = row % T;
  __shared__ float s[256];
  float a = 0.f;
  if (t < T - 1) {
    const float* px = x + (long long)(row + 1) * C;
    const float* ps = scan + (long long)row * C;
    for (int i = threadIdx.x; i < C; i += 256) { float d = px[i] - ps[i]; a += d * d; }
  }
  s[threadIdx.x] = a; __syncthreads();
  for (int st = 128; st > 0; st >>= 1) {
    if (threadIdx.x < st) s[threadIdx.x] += s[threadIdx.x + st];
    __syncthreads();
  }
  if (threadIdx.x == 0) gate[row] = (t < T - 1) ? s[0] : 0.f;
}

// per-batch deterministic top-k (largest, ties -> lowest index), sorted ascending
__global__ void __launch_bounds__(256) topk_kernel(const float* __restrict__ gate, int n, int k,
                                                   int* __restrict__ idx_out, float* __restrict__ g_out)
{
  __shared__ float sv[512];
  __shared__ float bv[256];
  __shared__ int   bi[256];
  __shared__ int   chosen[64];
  const float* gr = gate + (long long)blockIdx.x * n;
  for (int i = threadIdx.x; i < n; i += 256) sv[i] = gr[i];
  __syncthreads();
  for (int j = 0; j < k; ++j) {
    float best = -3.4e38f; int bidx = 0x7fffffff;
    for (int i = threadIdx.x; i < n; i += 256) {
      float v = sv[i];
      if (v > best || (v == best && i < bidx)) { best = v; bidx = i; }
    }
    bv[threadIdx.x] = best; bi[threadIdx.x] = bidx; __syncthreads();
    for (int s2 = 128; s2 > 0; s2 >>= 1) {
      if (threadIdx.x < s2) {
        float ov = bv[threadIdx.x + s2]; int oi = bi[threadIdx.x + s2];
        if (ov > bv[threadIdx.x] || (ov == bv[threadIdx.x] && oi < bi[threadIdx.x])) {
          bv[threadIdx.x] = ov; bi[threadIdx.x] = oi;
        }
      }
      __syncthreads();
    }
    if (threadIdx.x == 0) { chosen[j] = bi[0]; sv[bi[0]] = -3.4e38f; }
    __syncthreads();
  }
  if (threadIdx.x == 0) {
    for (int a2 = 1; a2 < k; ++a2) {           // insertion sort ascending, k<=64
      int key = chosen[a2]; int b2 = a2 - 1;
      while (b2 >= 0 && chosen[b2] > key) { chosen[b2 + 1] = chosen[b2]; --b2; }
      chosen[b2 + 1] = key;
    }
    for (int j = 0; j < k; ++j) {
      idx_out[blockIdx.x * k + j] = chosen[j];
      g_out[blockIdx.x * k + j]   = gr[chosen[j]];
    }
  }
}

__global__ void __launch_bounds__(256) gather_kernel(const float* __restrict__ src,
    const int* __restrict__ idx, const float* __restrict__ g,
    float* __restrict__ out, int T, int k, int C, long long n)
{
  long long i = (long long)blockIdx.x * 256 + threadIdx.x;
  if (i >= n) return;
  int c = (int)(i % C);
  long long br = i / C;
  int r = (int)(br % k);
  long long b = br / k;
  int j = idx[b * k + r];
  out[i] = src[((long long)b * T + j) * C + c] * g[b * k + r];
}

__global__ void __launch_bounds__(256) scatter_init_kernel(float* __restrict__ out,
    const float* __restrict__ pw, const float* __restrict__ pb, int S, int C, long long n)
{
  long long i = (long long)blockIdx.x * 256 + threadIdx.x;
  if (i >= n) return;
  int c = (int)(i % C);
  int r = (int)((i / C) % S);
  out[i] = pw[(long long)r * C + c] + pb[c];
}

__global__ void __launch_bounds__(256) scatter_add_kernel(float* __restrict__ out,
    const float* __restrict__ passed, const int* __restrict__ idx,
    const float* __restrict__ g, int S, int k, int C, long long n)
{
  long long i = (long long)blockIdx.x * 256 + threadIdx.x;
  if (i >= n) return;
  int c = (int)(i % C);
  long long br = i / C;
  int r = (int)(br % k);
  long long b = br / k;
  int j = idx[b * k + r];
  out[((long long)b * S + j) * C + c] += passed[i] * g[b * k + r];
}

__global__ void __launch_bounds__(256) pack_sg_kernel(const float* __restrict__ passed,
    float* __restrict__ out, int k, int C, int sdim, long long n)
{
  long long i = (long long)blockIdx.x * 256 + threadIdx.x;
  if (i >= n) return;
  int s2 = (int)(i % sdim);
  long long br = i / sdim;
  int r = (int)(br % k);
  long long b = br / k;
  out[i] = passed[((long long)b * k + r) * C + s2];
}

__global__ void __launch_bounds__(256) loss_row_kernel(const float* __restrict__ logits,
    const int* __restrict__ tgt, float* __restrict__ rowloss, int V)
{
  long long row = blockIdx.x;
  const float* p = logits + row * V;
  __shared__ float s[256];
  float mx = -3.4e38f;
  for (int i = threadIdx.x; i < V; i += 256) mx = fmaxf(mx, p[i]);
  s[threadIdx.x] = mx; __syncthreads();
  for (int st = 128; st > 0; st >>= 1) {
    if (threadIdx.x < st) s[threadIdx.x] = fmaxf(s[threadIdx.x], s[threadIdx.x + st]);
    __syncthreads();
  }
  mx = s[0]; __syncthreads();
  float sum = 0.f;
  for (int i = threadIdx.x; i < V; i += 256) sum += __expf(p[i] - mx);
  s[threadIdx.x] = sum; __syncthreads();
  for (int st = 128; st > 0; st >>= 1) {
    if (threadIdx.x < st) s[threadIdx.x] += s[threadIdx.x + st];
    __syncthreads();
  }
  if (threadIdx.x == 0) rowloss[row] = -(p[tgt[row]] - mx - logf(s[0]));
}

__global__ void __launch_bounds__(256) loss_reduce_kernel(const float* __restrict__ rl,
                                                          int R, float* __restrict__ out)
{
  __shared__ float s[256];
  float a = 0.f;
  for (int i = threadIdx.x; i < R; i += 256) a += rl[i];
  s[threadIdx.x] = a; __syncthreads();
  for (int st = 128; st > 0; st >>= 1) {
    if (threadIdx.x < st) s[threadIdx.x] += s[threadIdx.x + st];
    __syncthreads();
  }
  if (threadIdx.x == 0) out[0] = s[0] / R;
}

// ---------------- host side ----------------

namespace {

const int Bn = 4, Tn = 512, D1c = 64, D2c = 256, D3c = 1024;
const int K1c = 64, K2c = 16, NHc = 4, SDc = 10, Vc = 32000;

// flattened input indices (insertion order, depth-first)
enum {
  IN_IDX = 0, IN_TGT = 1,
  P_TOK = 2, P_POS = 3, P_HEADW = 4, P_HEADB = 5,
  L1_US_LN1 = 6, L1_US_QKV, L1_US_WO, L1_US_LN2, L1_US_FC, L1_US_PJ,
  L1_UPW = 12, L1_UPB, L1_POSW, L1_POSB,
  L1_DS_LN1 = 16, L1_DS_LN2, L1_DS_LN3, L1_DS_WQ, L1_DS_WK, L1_DS_WV, L1_DS_WC, L1_DS_FC, L1_DS_PJ,
  L1_DNW = 25, L1_DNB = 26,
  L1_DSC_LN1 = 27, L1_DSC_QKV, L1_DSC_WO, L1_DSC_LN2, L1_DSC_FC, L1_DSC_PJ,
  L1_SGW = 33, L1_SGB = 34,
  L2_US_LN1 = 35, L2_US_QKV, L2_US_WO, L2_US_LN2, L2_US_FC, L2_US_PJ,
  L2_UPW = 41, L2_UPB, L2_POSW, L2_POSB,
  L2_DS_LN1 = 45, L2_DS_LN2, L2_DS_LN3, L2_DS_WQ, L2_DS_WK, L2_DS_WV, L2_DS_WC, L2_DS_FC, L2_DS_PJ,
  L2_DNW = 54, L2_DNB = 55,
  L2_DSC_LN1 = 56, L2_DSC_QKV, L2_DSC_WO, L2_DSC_LN2, L2_DSC_FC, L2_DSC_PJ,
  L2_SGW = 62, L2_SGB = 63,
  IB_LN1 = 64, IB_QKV, IB_WO, IB_LN2, IB_FC, IB_PJ
};

struct Temps { float *xn, *sxn, *qkv, *q, *k, *v, *att, *y, *h; };

inline unsigned blocks_for(long long n) { return (unsigned)((n + 255) / 256); }

void gemm(hipStream_t s, const float* A, const float* B, const float* bias,
          const float* Res, float* C, int M, int N, int K,
          int lda, int ldb, int ldc,
          long long sAb, long long sAh, long long sBb, long long sBh,
          long long sCb, long long sCh, int batch, int nh, float alpha, int flags)
{
  dim3 g((N + 63) / 64, (M + 63) / 64, batch * nh);
  wmma_gemm_kernel<<<g, 128, 0, s>>>(A, B, bias, Res, C, M, N, K, lda, ldb, ldc,
                                     sAb, sAh, sBb, sBh, sCb, sCh, nh, alpha, flags);
}

void ln_go(hipStream_t s, const float* x, const float* w, float* y, int rows, int C) {
  ln_kernel<<<rows, 256, 0, s>>>(x, w, y, C);
}
void sm_go(hipStream_t s, float* x, int R, int N, int t, int causal) {
  softmax_kernel<<<R, 256, 0, s>>>(x, N, t, causal);
}

// transformer block with causal attention, in-place on x (b,t,c)
void run_block(hipStream_t s, float* x, int bN, int t, int c,
               const float* ln1, const float* wqkv, const float* wo,
               const float* ln2, const float* fc, const float* pj, Temps& T)
{
  const int rows = bN * t, nh = NHc, hd = c / nh;
  ln_go(s, x, ln1, T.xn, rows, c);
  gemm(s, T.xn, wqkv, 0, 0, T.qkv, rows, 3 * c, c, c, 3 * c, 3 * c,
       0, 0, 0, 0, 0, 0, 1, 1, 1.f, 0);
  gemm(s, T.qkv, T.qkv + c, 0, 0, T.att, t, t, hd, 3 * c, 3 * c, t,
       (long long)t * 3 * c, hd, (long long)t * 3 * c, hd,
       (long long)nh * t * t, (long long)t * t, bN, nh,
       1.f / sqrtf((float)hd), GEMM_TRANSB);
  sm_go(s, T.att, bN * nh * t, t, t, 1);
  gemm(s, T.att, T.qkv + 2 * c, 0, 0, T.y, t, hd, t, t, 3 * c, c,
       (long long)nh * t * t, (long long)t * t, (long long)t * 3 * c, hd,
       (long long)t * c, hd, bN, nh, 1.f, 0);
  gemm(s, T.y, wo, 0, x, x, rows, c, c, c, c, c, 0, 0, 0, 0, 0, 0, 1, 1, 1.f, GEMM_RES);
  ln_go(s, x, ln2, T.xn, rows, c);
  gemm(s, T.xn, fc, 0, 0, T.h, rows, 4 * c, c, c, 4 * c, 4 * c,
       0, 0, 0, 0, 0, 0, 1, 1, 1.f, GEMM_GELU);
  gemm(s, T.h, pj, 0, x, x, rows, c, 4 * c, 4 * c, c, c, 0, 0, 0, 0, 0, 0, 1, 1, 1.f, GEMM_RES);
}

// cross-attention sblock: queries from sx (b,st,c), keys/values from xin (b,t,c); in-place on sx
void run_sblock(hipStream_t s, const float* xin, float* sx, int bN, int t, int st, int c,
                const float* ln1, const float* ln2, const float* ln3,
                const float* wq, const float* wk, const float* wv, const float* wc,
                const float* fc, const float* pj, Temps& T)
{
  const int nh = NHc, hd = c / nh;
  ln_go(s, xin, ln1, T.xn, bN * t, c);
  ln_go(s, sx,  ln2, T.sxn, bN * st, c);
  gemm(s, T.sxn, wq, 0, 0, T.q, bN * st, c, c, c, c, c, 0, 0, 0, 0, 0, 0, 1, 1, 1.f, 0);
  gemm(s, T.xn,  wk, 0, 0, T.k, bN * t,  c, c, c, c, c, 0, 0, 0, 0, 0, 0, 1, 1, 1.f, 0);
  gemm(s, T.xn,  wv, 0, 0, T.v, bN * t,  c, c, c, c, c, 0, 0, 0, 0, 0, 0, 1, 1, 1.f, 0);
  gemm(s, T.q, T.k, 0, 0, T.att, st, t, hd, c, c, t,
       (long long)st * c, hd, (long long)t * c, hd,
       (long long)nh * st * t, (long long)st * t, bN, nh, 1.f, GEMM_TRANSB);
  sm_go(s, T.att, bN * nh * st, t, t, 0);
  gemm(s, T.att, T.v, 0, 0, T.y, st, hd, t, t, c, c,
       (long long)nh * st * t, (long long)st * t, (long long)t * c, hd,
       (long long)st * c, hd, bN, nh, 1.f, 0);
  gemm(s, T.y, wc, 0, sx, sx, bN * st, c, c, c, c, c, 0, 0, 0, 0, 0, 0, 1, 1, 1.f, GEMM_RES);
  ln_go(s, sx, ln3, T.sxn, bN * st, c);
  gemm(s, T.sxn, fc, 0, 0, T.h, bN * st, 4 * c, c, c, 4 * c, 4 * c,
       0, 0, 0, 0, 0, 0, 1, 1, 1.f, GEMM_GELU);
  gemm(s, T.h, pj, 0, sx, sx, bN * st, c, 4 * c, 4 * c, c, c,
       0, 0, 0, 0, 0, 0, 1, 1, 1.f, GEMM_RES);
}

} // namespace

extern "C" void kernel_launch(void* const* d_in, const int* in_sizes, int n_in,
                              void* d_out, int out_size, void* d_ws, size_t ws_size,
                              hipStream_t stream)
{
  (void)in_sizes; (void)n_in; (void)ws_size;
  const int* idx_in = (const int*)d_in[IN_IDX];
  const int* tgt_in = (const int*)d_in[IN_TGT];
  const float* P[70];
  for (int i = 2; i < 70; ++i) P[i] = (const float*)d_in[i];

  // bump allocator over workspace
  float* W = (float*)d_ws;
  size_t off = 0;
  auto alloc = [&](size_t n) { float* p = W + off; off += n; return p; };

  float* X     = alloc(131072);   // (4,512,64)
  float* SCAN1 = alloc(131072);
  float* SS1   = alloc(131072);
  float* GATE1 = alloc(2048);
  float* G1    = alloc(16384);    // (4,64,64)
  float* PU    = alloc(65536);    // (4,64,256)
  float* SCAN2 = alloc(65536);
  float* SS2   = alloc(65536);
  float* GATE2 = alloc(256);
  float* G2    = alloc(16384);    // (4,16,256)
  float* PU2   = alloc(65536);    // (4,16,1024)
  float* PS2   = alloc(65536);
  float* SGI2  = alloc(640);
  float* SG2   = alloc(256);
  float* SCAT2 = alloc(262144);   // (4,64,1024)
  float* PASS1 = alloc(65536);    // (4,64,256)
  float* SGI1  = alloc(2560);
  float* SG1   = alloc(2048);
  float* SCAT1 = alloc(524288);   // (4,512,256)
  float* XF    = alloc(131072);
  float* RLOSS = alloc(2048);
  int* idx1  = (int*)alloc(256);
  int* idx2  = (int*)alloc(64);
  int* idx2b = (int*)alloc(64);
  int* idx1b = (int*)alloc(256);
  float* gv1 = alloc(256);
  float* gv2 = alloc(64);
  float* gv3 = alloc(64);
  float* gv4 = alloc(256);

  Temps T;
  T.xn  = alloc(131072);
  T.sxn = alloc(524288);
  T.qkv = alloc(393216);
  T.q   = alloc(524288);
  T.k   = alloc(65536);
  T.v   = alloc(65536);
  T.att = alloc(4194304);
  T.y   = alloc(524288);
  T.h   = alloc(2097152);

  // 1) embedding
  long long nE = (long long)Bn * Tn * D1c;
  embed_kernel<<<blocks_for(nE), 256, 0, stream>>>(idx_in, P[P_TOK], P[P_POS], X, Tn, D1c, nE);

  // 2) L1 up_scan
  copy_kernel<<<blocks_for(nE), 256, 0, stream>>>(SCAN1, X, nE);
  run_block(stream, SCAN1, Bn, Tn, D1c,
            P[L1_US_LN1], P[L1_US_QKV], P[L1_US_WO], P[L1_US_LN2], P[L1_US_FC], P[L1_US_PJ], T);

  // 3) L1 gate + softmaxes + top-k gather
  gate_kernel<<<Bn * Tn, 256, 0, stream>>>(X, SCAN1, GATE1, Tn, D1c);
  sm_go(stream, GATE1, Bn, Tn, Tn, 0);
  copy_kernel<<<blocks_for(nE), 256, 0, stream>>>(SS1, SCAN1, nE);
  sm_go(stream, SS1, Bn * Tn, D1c, D1c, 0);
  topk_kernel<<<Bn, 256, 0, stream>>>(GATE1, Tn, K1c, idx1, gv1);
  long long nG1 = (long long)Bn * K1c * D1c;
  gather_kernel<<<blocks_for(nG1), 256, 0, stream>>>(SS1, idx1, gv1, G1, Tn, K1c, D1c, nG1);

  // 4) p_up = gathered @ up_w + up_b  -> (4,64,256)
  gemm(stream, G1, P[L1_UPW], P[L1_UPB], 0, PU, Bn * K1c, D2c, D1c, D1c, D2c, D2c,
       0, 0, 0, 0, 0, 0, 1, 1, 1.f, GEMM_BIAS);

  // ---- L2 patcher ----
  long long nPU = (long long)Bn * K1c * D2c;
  copy_kernel<<<blocks_for(nPU), 256, 0, stream>>>(SCAN2, PU, nPU);
  run_block(stream, SCAN2, Bn, K1c, D2c,
            P[L2_US_LN1], P[L2_US_QKV], P[L2_US_WO], P[L2_US_LN2], P[L2_US_FC], P[L2_US_PJ], T);
  gate_kernel<<<Bn * K1c, 256, 0, stream>>>(PU, SCAN2, GATE2, K1c, D2c);
  sm_go(stream, GATE2, Bn, K1c, K1c, 0);
  copy_kernel<<<blocks_for(nPU), 256, 0, stream>>>(SS2, SCAN2, nPU);
  sm_go(stream, SS2, Bn * K1c, D2c, D2c, 0);
  topk_kernel<<<Bn, 256, 0, stream>>>(GATE2, K1c, K2c, idx2, gv2);
  long long nG2 = (long long)Bn * K2c * D2c;
  gather_kernel<<<blocks_for(nG2), 256, 0, stream>>>(SS2, idx2, gv2, G2, K1c, K2c, D2c, nG2);
  gemm(stream, G2, P[L2_UPW], P[L2_UPB], 0, PU2, Bn * K2c, D3c, D2c, D2c, D3c, D3c,
       0, 0, 0, 0, 0, 0, 1, 1, 1.f, GEMM_BIAS);

  // inner block (D3, t=16, causal)
  long long nPU2 = (long long)Bn * K2c * D3c;
  copy_kernel<<<blocks_for(nPU2), 256, 0, stream>>>(PS2, PU2, nPU2);
  run_block(stream, PS2, Bn, K2c, D3c,
            P[IB_LN1], P[IB_QKV], P[IB_WO], P[IB_LN2], P[IB_FC], P[IB_PJ], T);

  // L2 scatter gate + scatter + down_scatter sblock
  long long nSGI2 = (long long)Bn * K2c * SDc;
  pack_sg_kernel<<<blocks_for(nSGI2), 256, 0, stream>>>(PS2, SGI2, K2c, D3c, SDc, nSGI2);
  gemm(stream, SGI2, P[L2_SGW], P[L2_SGB], 0, SG2, Bn, K1c, K2c * SDc,
       K2c * SDc, K1c, K1c, 0, 0, 0, 0, 0, 0, 1, 1, 1.f, GEMM_BIAS);
  sm_go(stream, SG2, Bn, K1c, K1c, 0);
  topk_kernel<<<Bn, 256, 0, stream>>>(SG2, K1c, K2c, idx2b, gv3);
  long long nSC2 = (long long)Bn * K1c * D3c;
  scatter_init_kernel<<<blocks_for(nSC2), 256, 0, stream>>>(SCAT2, P[L2_POSW], P[L2_POSB], K1c, D3c, nSC2);
  scatter_add_kernel<<<blocks_for(nPU2), 256, 0, stream>>>(SCAT2, PS2, idx2b, gv3, K1c, K2c, D3c, nPU2);
  run_sblock(stream, PS2, SCAT2, Bn, K2c, K1c, D3c,
             P[L2_DS_LN1], P[L2_DS_LN2], P[L2_DS_LN3], P[L2_DS_WQ], P[L2_DS_WK],
             P[L2_DS_WV], P[L2_DS_WC], P[L2_DS_FC], P[L2_DS_PJ], T);

  // p_down2 + scan2 residual -> PASS1, then L2 down_scan
  gemm(stream, SCAT2, P[L2_DNW], P[L2_DNB], SCAN2, PASS1, Bn * K1c, D2c, D3c,
       D3c, D2c, D2c, 0, 0, 0, 0, 0, 0, 1, 1, 1.f, GEMM_BIAS | GEMM_RES);
  run_block(stream, PASS1, Bn, K1c, D2c,
            P[L2_DSC_LN1], P[L2_DSC_QKV], P[L2_DSC_WO], P[L2_DSC_LN2], P[L2_DSC_FC], P[L2_DSC_PJ], T);

  // ---- back in L1 ----
  long long nSGI1 = (long long)Bn * K1c * SDc;
  pack_sg_kernel<<<blocks_for(nSGI1), 256, 0, stream>>>(PASS1, SGI1, K1c, D2c, SDc, nSGI1);
  gemm(stream, SGI1, P[L1_SGW], P[L1_SGB], 0, SG1, Bn, Tn, K1c * SDc,
       K1c * SDc, Tn, Tn, 0, 0, 0, 0, 0, 0, 1, 1, 1.f, GEMM_BIAS);
  sm_go(stream, SG1, Bn, Tn, Tn, 0);
  topk_kernel<<<Bn, 256, 0, stream>>>(SG1, Tn, K1c, idx1b, gv4);
  long long nSC1 = (long long)Bn * Tn * D2c;
  scatter_init_kernel<<<blocks_for(nSC1), 256, 0, stream>>>(SCAT1, P[L1_POSW], P[L1_POSB], Tn, D2c, nSC1);
  long long nP1 = (long long)Bn * K1c * D2c;
  scatter_add_kernel<<<blocks_for(nP1), 256, 0, stream>>>(SCAT1, PASS1, idx1b, gv4, Tn, K1c, D2c, nP1);
  run_sblock(stream, PASS1, SCAT1, Bn, K1c, Tn, D2c,
             P[L1_DS_LN1], P[L1_DS_LN2], P[L1_DS_LN3], P[L1_DS_WQ], P[L1_DS_WK],
             P[L1_DS_WV], P[L1_DS_WC], P[L1_DS_FC], P[L1_DS_PJ], T);

  // p_down + scan1 residual -> XF, then L1 down_scan
  gemm(stream, SCAT1, P[L1_DNW], P[L1_DNB], SCAN1, XF, Bn * Tn, D1c, D2c,
       D2c, D1c, D1c, 0, 0, 0, 0, 0, 0, 1, 1, 1.f, GEMM_BIAS | GEMM_RES);
  run_block(stream, XF, Bn, Tn, D1c,
            P[L1_DSC_LN1], P[L1_DSC_QKV], P[L1_DSC_WO], P[L1_DSC_LN2], P[L1_DSC_FC], P[L1_DSC_PJ], T);

  // logits straight into d_out (memory floor of the whole net)
  float* logits = (float*)d_out;
  gemm(stream, XF, P[P_HEADW], P[P_HEADB], 0, logits, Bn * Tn, Vc, D1c,
       D1c, Vc, Vc, 0, 0, 0, 0, 0, 0, 1, 1, 1.f, GEMM_BIAS);

  // cross-entropy: per-row logsumexp, then deterministic tree reduction
  loss_row_kernel<<<Bn * Tn, 256, 0, stream>>>(logits, tgt_in, RLOSS, Vc);
  loss_reduce_kernel<<<1, 256, 0, stream>>>(RLOSS, Bn * Tn, logits + ((long long)out_size - 1));
}
